// GAT_3109556322911
// MI455X (gfx1250) — matrix operand order repaired and checked by reading, not compiled, not run on hardware
//
#include <hip/hip_runtime.h>
#include <hip/hip_fp16.h>

typedef _Float16 v16h __attribute__((ext_vector_type(16)));
typedef float    v8f  __attribute__((ext_vector_type(8)));

static constexpr int FIN   = 128;  // F_IN and HEADS*HID
static constexpr int HID   = 32;
static constexpr int HEADS = 4;
static constexpr int NCLS  = 40;

__device__ __forceinline__ void atomicMaxF(float* addr, float val) {
    // IEEE float max via native integer atomics (init value must be -inf)
    if (val >= 0.0f) atomicMax((int*)addr, __float_as_int(val));
    else             atomicMin((unsigned int*)addr, __float_as_uint(val));
}

// ---------------------------------------------------------------- conversions
__global__ void k_f32_to_f16(const float* __restrict__ in,
                             _Float16* __restrict__ out, int n) {
    int i = blockIdx.x * blockDim.x + threadIdx.x;
    if (i < n) out[i] = (_Float16)in[i];
}

// ---------------------------------------------------------------- WMMA GEMM
// O[M x (NCT*16)] = A[M x 128] (f16) @ W[128 x (NCT*16)] (f16), f32 accumulate.
// One wave per 16x16 output tile; wave id within block = column tile.
template <int NCT>
__global__ void __launch_bounds__(NCT * 32)
k_wmma_gemm(const _Float16* __restrict__ A, const _Float16* __restrict__ W,
            float* __restrict__ O, int M) {
    constexpr int NC = NCT * 16;
    constexpr int K  = 128;
    const int wave    = threadIdx.x >> 5;   // column tile
    const int lane    = threadIdx.x & 31;
    const int r0      = blockIdx.x * 16;
    const int colBase = wave * 16;
    const int row     = lane & 15;
    const int g       = lane >> 4;          // K-half select for A fragment
    if (r0 >= M) return;

    v8f acc = {};
    const _Float16* arow = A + (size_t)(r0 + row) * K;
#pragma unroll
    for (int kk = 0; kk < K / 32; ++kk) {
        v16h a, b;
        // A 16x32 f16 fragment: lane holds row, halves 0..7 = K ka..ka+7,
        // halves 8..15 = K ka+16..ka+23  (two contiguous 16B loads)
        const _Float16* ap = arow + kk * 32 + g * 8;
        ((float4*)&a)[0] = *(const float4*)(ap);
        ((float4*)&a)[1] = *(const float4*)(ap + 16);
        // B 32x16 f16 fragment: lane holds col colBase+row, halves 0..7 = K ka..ka+7,
        // halves 8..15 = K ka+16..ka+23 (strided by NC)
        const _Float16* bp = W + (size_t)(kk * 32 + g * 8) * NC + colBase + row;
#pragma unroll
        for (int s = 0; s < 8; ++s) {
            b[s]     = bp[(size_t)s * NC];
            b[8 + s] = bp[(size_t)(16 + s) * NC];
        }
        acc = __builtin_amdgcn_wmma_f32_16x16x32_f16(
            /*neg_a=*/false, a, /*neg_b=*/false, b,
            /*c_mod=*/(short)0, acc, /*reuse_a=*/false, /*reuse_b=*/false);
    }
    // D layout: lane -> col = lane&15, VGPR v -> row = v + 8*(lane>>4)
    float* out = O + (size_t)r0 * NC + colBase + (lane & 15);
#pragma unroll
    for (int v = 0; v < 8; ++v)
        out[(size_t)(v + 8 * g) * NC] = acc[v];
}

// ---------------------------------------------------------------- attention
// alpha_s / alpha_d: per (node, head) dot of 32 channels. t*32 == offset of
// h[n, head, :] since t = n*H + head.
__global__ void k_alpha(const float* __restrict__ h, const float* __restrict__ a_s,
                        const float* __restrict__ a_d, float* __restrict__ AS,
                        float* __restrict__ AD, int NH, int H) {
    int t = blockIdx.x * blockDim.x + threadIdx.x;
    if (t >= NH) return;
    int hd = t % H;
    const float4* hv = (const float4*)(h + (size_t)t * 32);
    const float4* sv = (const float4*)(a_s + hd * 32);
    const float4* dv = (const float4*)(a_d + hd * 32);
    float ss = 0.f, dd = 0.f;
#pragma unroll
    for (int j = 0; j < 8; ++j) {
        float4 x = hv[j], s = sv[j], d = dv[j];
        ss += x.x * s.x + x.y * s.y + x.z * s.z + x.w * s.w;
        dd += x.x * d.x + x.y * d.y + x.z * d.z + x.w * d.w;
    }
    AS[t] = ss;
    AD[t] = dd;
}

__global__ void k_init(float* __restrict__ acc, float* __restrict__ m,
                       float* __restrict__ den, int NHC, int NH) {
    int i = blockIdx.x * blockDim.x + threadIdx.x;
    if (i < NHC) acc[i] = 0.f;
    if (i < NH) { m[i] = -__builtin_inff(); den[i] = 0.f; }
}

// pass 1: segment max over incoming edges (incl. self-loops)
__global__ void k_edge_max(const long long* __restrict__ ei, int E, int T,
                           const float* __restrict__ AS, const float* __restrict__ AD,
                           float* __restrict__ m, int H) {
    int e = blockIdx.x * blockDim.x + threadIdx.x;
    if (e >= T) return;
    int s, d;
    if (e < E) { s = (int)ei[e]; d = (int)ei[(size_t)E + e]; }
    else       { s = d = e - E; }
    for (int hd = 0; hd < H; ++hd) {
        float v = AS[s * H + hd] + AD[d * H + hd];
        v = v > 0.f ? v : 0.2f * v;          // leaky relu
        atomicMaxF(&m[d * H + hd], v);
    }
}

// pass 2: one wave per edge. Accumulate den[d,h] += e and acc[d,:] += e*h[s,:]
// (normalization deferred to k_finalize: out = acc/den == softmax-weighted sum)
template <int H, int C>
__global__ void __launch_bounds__(256)
k_edge_agg(const long long* __restrict__ ei, int E, int T,
           const float* __restrict__ h, const float* __restrict__ AS,
           const float* __restrict__ AD, const float* __restrict__ m,
           float* __restrict__ den, float* __restrict__ acc) {
    constexpr int HC = H * C;
    constexpr int PL = HC / 32;              // channels per lane
    int w    = (blockIdx.x * blockDim.x + threadIdx.x) >> 5;
    int lane = threadIdx.x & 31;
    if (w >= T) return;
    int s, d;
    if (w < E) { s = (int)ei[w]; d = (int)ei[(size_t)E + w]; }
    else       { s = d = w - E; }
    int c0 = lane * PL;
    int hd = c0 / C;
    float v = AS[s * H + hd] + AD[d * H + hd];
    v = v > 0.f ? v : 0.2f * v;
    float p = __expf(v - m[d * H + hd]);
    if ((c0 % C) == 0) unsafeAtomicAdd(&den[d * H + hd], p);
    const float* hs = h + (size_t)s * HC + c0;
    float* od = acc + (size_t)d * HC + c0;
    if constexpr (PL == 4) {
        float4 hv = *(const float4*)hs;
        unsafeAtomicAdd(od + 0, hv.x * p);
        unsafeAtomicAdd(od + 1, hv.y * p);
        unsafeAtomicAdd(od + 2, hv.z * p);
        unsafeAtomicAdd(od + 3, hv.w * p);
    } else {
        unsafeAtomicAdd(od, hs[0] * p);
    }
}

// normalize, +bias, relu; write f16 (next GEMM input) or f32 (final features)
template <bool F16OUT>
__global__ void k_finalize(const float* __restrict__ acc, const float* __restrict__ den,
                           const float* __restrict__ bias, _Float16* __restrict__ o16,
                           float* __restrict__ o32, int NHC, int H, int HC) {
    int i = blockIdx.x * blockDim.x + threadIdx.x;
    if (i >= NHC) return;
    int c  = i % HC;
    int hd = c / 32;
    int n  = i / HC;
    float v = acc[i] / (den[n * H + hd] + 1e-16f) + bias[c];
    v = v > 0.f ? v : 0.f;
    if constexpr (F16OUT) o16[i] = (_Float16)v;
    else                  o32[i] = v;
}

__global__ void k_linear(const float* __restrict__ x, const float* __restrict__ w,
                         const float* __restrict__ b, float* __restrict__ out, int NT) {
    int t = blockIdx.x * blockDim.x + threadIdx.x;
    if (t >= NT) return;
    int n = t / NCLS, k = t % NCLS;
    const float* xp = x + (size_t)n * HID;
    float s = b[k];
#pragma unroll
    for (int c = 0; c < HID; ++c) s += xp[c] * w[c * NCLS + k];
    out[t] = s;
}

// ---------------------------------------------------------------- launcher
extern "C" void kernel_launch(void* const* d_in, const int* in_sizes, int n_in,
                              void* d_out, int out_size, void* d_ws, size_t ws_size,
                              hipStream_t stream) {
    const float*     x     = (const float*)d_in[0];
    const long long* ei    = (const long long*)d_in[1];   // int64 [2,E]
    const float*     W0    = (const float*)d_in[2];
    const float*     as0   = (const float*)d_in[3];
    const float*     ad0   = (const float*)d_in[4];
    const float*     b0    = (const float*)d_in[5];
    const float*     W1    = (const float*)d_in[6];
    const float*     as1   = (const float*)d_in[7];
    const float*     ad1   = (const float*)d_in[8];
    const float*     b1    = (const float*)d_in[9];
    const float*     W2    = (const float*)d_in[10];
    const float*     as2   = (const float*)d_in[11];
    const float*     ad2   = (const float*)d_in[12];
    const float*     b2    = (const float*)d_in[13];
    const float*     lin_w = (const float*)d_in[14];
    const float*     lin_b = (const float*)d_in[15];

    const int N = in_sizes[0] / FIN;
    const int E = in_sizes[1] / 2;
    const int T = E + N;                      // edges + self-loops

    char* base = (char*)d_ws;
    size_t off = 0;
    auto carve = [&](size_t bytes) {
        void* p = base + off;
        off = (off + bytes + 255) & ~(size_t)255;
        return p;
    };
    _Float16* A16 = (_Float16*)carve((size_t)N * FIN * 2);   // GEMM A (f16)
    float*    Hb  = (float*)carve((size_t)N * FIN * 4);      // GEMM output h
    float*    ACC = (float*)carve((size_t)N * FIN * 4);      // scatter accumulator
    float*    AS  = (float*)carve((size_t)N * HEADS * 4);
    float*    AD  = (float*)carve((size_t)N * HEADS * 4);
    float*    MX  = (float*)carve((size_t)N * HEADS * 4);
    float*    DEN = (float*)carve((size_t)N * HEADS * 4);
    float*    X2  = (float*)carve((size_t)N * HID * 4);      // layer-2 activation
    _Float16* W16 = (_Float16*)carve((size_t)FIN * FIN * 2); // converted weight
    (void)ws_size; (void)n_in; (void)out_size;

    auto cdiv = [](long long a, long long b) { return (int)((a + b - 1) / b); };

    // x -> f16 once
    k_f32_to_f16<<<cdiv((long long)N * FIN, 256), 256, 0, stream>>>(x, A16, N * FIN);

    const float* Ws[2]  = {W0, W1};
    const float* asl[2] = {as0, as1};
    const float* adl[2] = {ad0, ad1};
    const float* bl[2]  = {b0, b1};
    for (int L = 0; L < 2; ++L) {
        k_f32_to_f16<<<cdiv(FIN * FIN, 256), 256, 0, stream>>>(Ws[L], W16, FIN * FIN);
        k_wmma_gemm<8><<<N / 16, 256, 0, stream>>>(A16, W16, Hb, N);
        k_alpha<<<cdiv((long long)N * HEADS, 256), 256, 0, stream>>>(
            Hb, asl[L], adl[L], AS, AD, N * HEADS, HEADS);
        k_init<<<cdiv((long long)N * FIN, 256), 256, 0, stream>>>(
            ACC, MX, DEN, N * FIN, N * HEADS);
        k_edge_max<<<cdiv(T, 256), 256, 0, stream>>>(ei, E, T, AS, AD, MX, HEADS);
        k_edge_agg<HEADS, HID><<<cdiv(T, 8), 256, 0, stream>>>(
            ei, E, T, Hb, AS, AD, MX, DEN, ACC);
        k_finalize<true><<<cdiv((long long)N * FIN, 256), 256, 0, stream>>>(
            ACC, DEN, bl[L], A16, nullptr, N * FIN, HEADS, FIN);
    }

    // layer 2: 128 -> 32, single head
    k_f32_to_f16<<<cdiv(FIN * HID, 256), 256, 0, stream>>>(W2, W16, FIN * HID);
    k_wmma_gemm<2><<<N / 16, 64, 0, stream>>>(A16, W16, Hb, N);
    k_alpha<<<cdiv(N, 256), 256, 0, stream>>>(Hb, as2, ad2, AS, AD, N, 1);
    k_init<<<cdiv((long long)N * HID, 256), 256, 0, stream>>>(ACC, MX, DEN, N * HID, N);
    k_edge_max<<<cdiv(T, 256), 256, 0, stream>>>(ei, E, T, AS, AD, MX, 1);
    k_edge_agg<1, HID><<<cdiv(T, 8), 256, 0, stream>>>(ei, E, T, Hb, AS, AD, MX, DEN, ACC);
    k_finalize<false><<<cdiv((long long)N * HID, 256), 256, 0, stream>>>(
        ACC, DEN, b2, nullptr, X2, N * HID, 1, HID);

    // final linear 32 -> 40
    k_linear<<<cdiv((long long)N * NCLS, 256), 256, 0, stream>>>(
        X2, lin_w, lin_b, (float*)d_out, N * NCLS);
}
